// VQ_Payam_GSOFT_8821862826425
// MI455X (gfx1250) — matrix operand order repaired and checked by reading, not compiled, not run on hardware
//
#include <hip/hip_runtime.h>
#include <math.h>

typedef __attribute__((ext_vector_type(16))) _Float16 v16h;
typedef __attribute__((ext_vector_type(8)))  _Float16 v8h;
typedef __attribute__((ext_vector_type(8)))  float    v8f;
typedef unsigned int u32x4 __attribute__((ext_vector_type(4)));
typedef int          i32x8 __attribute__((ext_vector_type(8)));
typedef int          i32x4 __attribute__((ext_vector_type(4)));

#ifndef __has_builtin
#define __has_builtin(x) 0
#endif
#if __has_builtin(__builtin_amdgcn_tensor_load_to_lds) && __has_builtin(__builtin_amdgcn_s_wait_tensorcnt)
#define HAVE_TDM 1
#else
#define HAVE_TDM 0
#endif

#define N_TOK     131072
#define D_DIM     128
#define K_CODE    512
#define TILE_M    16
#define N_TILES   (N_TOK / TILE_M)   // 8192
#define GRID_BLKS 512
#define BLK       256
#define LOG_K     6.238324625039508f // ln(512)

// LDS strides (elements). 136 halves = 272B -> 16B-aligned rows for b128 frags,
// and 272/4=68 dwords -> bank rotation of 4 per row (no 16-way conflicts).
#define EMB_STRIDE 136   // f16 codebook rows
#define XT_STRIDE  136   // f16 x-tile rows
#define EA_STRIDE  520   // f16 encodings tile rows (1040B, 16B-aligned)
#define LG_STRIDE  516   // f32 logits tile rows

// output layout: [KL][quantized N*D][perplexity][encodings N*K]
#define Q_OFF ((size_t)1)
#define P_OFF ((size_t)1 + (size_t)N_TOK * (size_t)D_DIM)
#define E_OFF (P_OFF + 1)

// ---- Tensor Data Mover: 2D tile load (data_size=4B), groups per ISA 8.3/8.4
// clang-23 / therock lane: 6-arg builtin (u32x4, i32x8, i32x4, i32x4, i32x8, i32 cpol)
__device__ __forceinline__ void tdm_load_2d(const void* gptr, unsigned lds_byte_off,
                                            unsigned dim0, unsigned tile1)
{
#if HAVE_TDM
    const unsigned long long ga = (unsigned long long)(uintptr_t)gptr;
    u32x4 g0;
    g0[0] = 1u;                                   // count=1, user descriptor
    g0[1] = lds_byte_off;                         // lds_addr
    g0[2] = (unsigned)(ga & 0xffffffffu);         // global_addr[31:0]
    g0[3] = (unsigned)((ga >> 32) & 0x01ffffffu)  // global_addr[56:32]
            | 0x80000000u;                        // type=2 ("image")
    i32x8 g1;
    g1[0] = (int)(2u << 16);                      // data_size=4B (code 2)
    g1[1] = (int)(dim0 << 16);                    // tensor_dim0[15:0]
    g1[2] = (int)((dim0 >> 16) | (((unsigned)N_TOK & 0xffffu) << 16)); // dim0 hi | dim1 lo
    g1[3] = (int)((((unsigned)N_TOK) >> 16) | (dim0 << 16));           // dim1 hi | tile_dim0
    g1[4] = (int)tile1;                           // tile_dim1 (tile_dim2=0)
    g1[5] = (int)dim0;                            // tensor_dim0_stride[31:0]
    g1[6] = 0;                                    // stride hi / dim1_stride lo
    g1[7] = 0;
    const i32x4 z4 = {0, 0, 0, 0};                // groups 2/3 unused (2D tensor)
    const i32x8 z8 = {0, 0, 0, 0, 0, 0, 0, 0};
    __builtin_amdgcn_tensor_load_to_lds(g0, g1, z4, z4, z8, 0);
#else
    (void)gptr; (void)lds_byte_off; (void)dim0; (void)tile1;
#endif
}

__device__ __forceinline__ v16h frag_from_2x8(const _Float16* lo, const _Float16* hi)
{
    const v8h a = *(const v8h*)lo;
    const v8h b = *(const v8h*)hi;
    return __builtin_shufflevector(a, b, 0, 1, 2, 3, 4, 5, 6, 7,
                                         8, 9, 10, 11, 12, 13, 14, 15);
}

__global__ __launch_bounds__(BLK)
void vq_fused_kernel(const float* __restrict__ x,
                     const float* __restrict__ u,
                     const float* __restrict__ emb,
                     float* __restrict__ out,
                     float* __restrict__ ws)
{
    // ~273KB LDS: legal only on CDNA5 (320KB/WGP)
    __shared__ _Float16 emb_lds[K_CODE * EMB_STRIDE];     // 139,264 B
    __shared__ _Float16 xt_lds [TILE_M * XT_STRIDE];      //   4,352 B
    __shared__ _Float16 encA   [TILE_M * EA_STRIDE];      //  16,640 B
    __shared__ float    lgt    [TILE_M * LG_STRIDE];      //  33,024 B
    __shared__ float    x_stage[2 * TILE_M * D_DIM];      //  16,384 B (double buffered)
    __shared__ float    u_stage[2 * TILE_M * K_CODE];     //  65,536 B (double buffered)
    __shared__ float    esq    [K_CODE];
    __shared__ float    xsq    [TILE_M];
    __shared__ float    avg    [K_CODE];

    const int tid  = threadIdx.x;
    const int lane = tid & 31;
    const int wave = tid >> 5;

    // ---- one-time: codebook -> LDS (f16), ||e||^2, zero usage accumulator
    for (int idx = tid; idx < K_CODE * 32; idx += BLK) {
        const int row = idx >> 5, c4 = idx & 31;
        const float4 v = ((const float4*)emb)[row * 32 + c4];
        _Float16* dst = &emb_lds[row * EMB_STRIDE + c4 * 4];
        dst[0] = (_Float16)v.x; dst[1] = (_Float16)v.y;
        dst[2] = (_Float16)v.z; dst[3] = (_Float16)v.w;
    }
    for (int k = tid; k < K_CODE; k += BLK) avg[k] = 0.0f;
    __syncthreads();
    for (int k = tid; k < K_CODE; k += BLK) {
        float s = 0.0f;
        const _Float16* r = &emb_lds[k * EMB_STRIDE];
        for (int dd = 0; dd < D_DIM; ++dd) { const float e = (float)r[dd]; s += e * e; }
        esq[k] = s;
    }
    float kl_local = 0.0f;
    __syncthreads();

    // ---- issue TDM for the first tile (wave 0 only; EXEC-independent DMA)
    int buf = 0;
    if (wave == 0) {
        const int t0 = blockIdx.x * TILE_M;
        tdm_load_2d(x + (size_t)t0 * D_DIM,  (unsigned)(uintptr_t)&x_stage[0], D_DIM,  TILE_M);
        tdm_load_2d(u + (size_t)t0 * K_CODE, (unsigned)(uintptr_t)&u_stage[0], K_CODE, TILE_M);
    }

    for (int tile = blockIdx.x; tile < N_TILES; tile += gridDim.x) {
        const int t0 = tile * TILE_M;
        float* xs = &x_stage[buf * TILE_M * D_DIM];
        float* us = &u_stage[buf * TILE_M * K_CODE];

#if HAVE_TDM
        if (wave == 0) __builtin_amdgcn_s_wait_tensorcnt(0);
#else
        // fallback: cooperative staging copies (same structure, no TDM)
        for (int idx = tid; idx < TILE_M * (D_DIM / 4); idx += BLK)
            ((float4*)xs)[idx] = ((const float4*)(x + (size_t)t0 * D_DIM))[idx];
        for (int idx = tid; idx < TILE_M * (K_CODE / 4); idx += BLK)
            ((float4*)us)[idx] = ((const float4*)(u + (size_t)t0 * K_CODE))[idx];
#endif
        __syncthreads();   // staged x/u visible to all waves

        // convert x tile f32 -> f16 fragment layout + ||x||^2 (from f32)
        for (int idx = tid; idx < TILE_M * D_DIM; idx += BLK) {
            const int row = idx >> 7, c = idx & 127;
            xt_lds[row * XT_STRIDE + c] = (_Float16)xs[idx];
        }
        if (tid < TILE_M) {
            float s = 0.0f;
            const float* r = &xs[tid * D_DIM];
            for (int dd = 0; dd < D_DIM; ++dd) s += r[dd] * r[dd];
            xsq[tid] = s;
        }
        __syncthreads();

        // prefetch NEXT tile via TDM into the other buffer (overlaps all compute below)
        const int ntile = tile + gridDim.x;
        if (wave == 0 && ntile < N_TILES) {
            const int nt0 = ntile * TILE_M;
            tdm_load_2d(x + (size_t)nt0 * D_DIM,
                        (unsigned)(uintptr_t)&x_stage[(buf ^ 1) * TILE_M * D_DIM], D_DIM, TILE_M);
            tdm_load_2d(u + (size_t)nt0 * K_CODE,
                        (unsigned)(uintptr_t)&u_stage[(buf ^ 1) * TILE_M * K_CODE], K_CODE, TILE_M);
        }

        // ---- stage 1: logits[16x512] = 2*x.E^T - |x|^2 - |e|^2 via WMMA f16
        {
            const int mrow = lane & 15;
            const int half = lane >> 4;          // 0 or 1
            for (int jt = 0; jt < 4; ++jt) {
                const int n0 = (wave * 4 + jt) * 16;
                v8f acc = {};
                for (int d = 0; d < D_DIM; d += 32) {
                    const int ka = d + half * 16;    // A: contiguous k = 16*half + s
                    const _Float16* xr = &xt_lds[mrow * XT_STRIDE];
                    const v16h a = frag_from_2x8(xr + ka, xr + ka + 8);
                    const int kb = d + half * 16;    // B: 16-bit 32x16 layout
                    const _Float16* er = &emb_lds[(n0 + mrow) * EMB_STRIDE + kb];
                    const v16h b = frag_from_2x8(er, er + 8);
                    acc = __builtin_amdgcn_wmma_f32_16x16x32_f16(
                        false, a, false, b, (short)0, acc, false, false);
                }
                const int mbase = half * 8;
#pragma unroll
                for (int r = 0; r < 8; ++r) {
                    const int m = mbase + r;
                    const int n = n0 + mrow;
                    lgt[m * LG_STRIDE + n] = 2.0f * acc[r] - xsq[m] - esq[n];
                }
            }
        }
        __syncthreads();

        // ---- stage 2a: per-row softmax / KL / gumbel-softmax (2 rows per wave)
        for (int rr = 0; rr < 2; ++rr) {
            const int m = wave * 2 + rr;
            float v[16];
#pragma unroll
            for (int i = 0; i < 16; ++i) v[i] = lgt[m * LG_STRIDE + i * 32 + lane];

            float mx = -3.0e38f;
#pragma unroll
            for (int i = 0; i < 16; ++i) mx = fmaxf(mx, v[i]);
            for (int s = 16; s > 0; s >>= 1) mx = fmaxf(mx, __shfl_xor(mx, s, 32));
            float se = 0.0f;
#pragma unroll
            for (int i = 0; i < 16; ++i) se += __expf(v[i] - mx);
            for (int s = 16; s > 0; s >>= 1) se += __shfl_xor(se, s, 32);
            const float lz = __logf(se);
#pragma unroll
            for (int i = 0; i < 16; ++i) {
                const float lp = v[i] - mx - lz;
                const float p  = __expf(lp);
                kl_local += p * (lp + LOG_K);   // exp underflow -> 0 term, matches masking
            }

            // gumbel from LDS-staged u, second softmax at tau=0.5 -> *2
            const float* ur = &us[m * K_CODE];
            float y[16];
#pragma unroll
            for (int i = 0; i < 16; ++i) {
                float uu = ur[i * 32 + lane];
                uu = fminf(fmaxf(uu, 1.17549435e-38f), 1.0f - 1e-7f);
                const float g = -__logf(-__logf(uu));
                y[i] = (v[i] + g) * 2.0f;
            }
            float mx2 = -3.0e38f;
#pragma unroll
            for (int i = 0; i < 16; ++i) mx2 = fmaxf(mx2, y[i]);
            for (int s = 16; s > 0; s >>= 1) mx2 = fmaxf(mx2, __shfl_xor(mx2, s, 32));
            float se2 = 0.0f;
#pragma unroll
            for (int i = 0; i < 16; ++i) se2 += __expf(y[i] - mx2);
            for (int s = 16; s > 0; s >>= 1) se2 += __shfl_xor(se2, s, 32);
            const float inv = 1.0f / se2;

            float* oe = &out[E_OFF + (size_t)(t0 + m) * K_CODE];
#pragma unroll
            for (int i = 0; i < 16; ++i) {
                const float e = __expf(y[i] - mx2) * inv;
                const int k = i * 32 + lane;
                oe[k] = e;                              // encodings output
                encA[m * EA_STRIDE + k] = (_Float16)e;  // A-matrix for stage 2b
                atomicAdd(&avg[k], e);                  // ds_add_f32 usage accumulation
            }
        }
        __syncthreads();

        // ---- stage 2b: quantized[16x128] = enc[16x512] @ E[512x128] via WMMA
        {
            const int mrow = lane & 15;
            const int half = lane >> 4;
            const int n0q  = wave * 16;   // D-columns tile per wave
            v8f q = {};
            for (int k = 0; k < K_CODE; k += 32) {
                const int ka = k + half * 16;
                const _Float16* er = &encA[mrow * EA_STRIDE];
                const v16h a = frag_from_2x8(er + ka, er + ka + 8);
                const int kb = k + half * 16;
                v16h b;
#pragma unroll
                for (int h = 0; h < 16; ++h) b[h] = emb_lds[(kb + h) * EMB_STRIDE + n0q + mrow];
                q = __builtin_amdgcn_wmma_f32_16x16x32_f16(
                    false, a, false, b, (short)0, q, false, false);
            }
            const int mbase = half * 8;
            float* oq = &out[Q_OFF];
#pragma unroll
            for (int r = 0; r < 8; ++r)
                oq[(size_t)(t0 + mbase + r) * D_DIM + n0q + mrow] = q[r];
        }
        __syncthreads();
        buf ^= 1;
    }

    // ---- flush block-local reductions
    for (int k = tid; k < K_CODE; k += BLK) atomicAdd(&ws[k], avg[k]);
    for (int s = 16; s > 0; s >>= 1) kl_local += __shfl_xor(kl_local, s, 32);
    if (lane == 0) atomicAdd(&ws[K_CODE], kl_local);
}

__global__ void vq_finalize_kernel(const float* __restrict__ ws, float* __restrict__ out)
{
    __shared__ float red[K_CODE];
    const int k = threadIdx.x;
    const float a = ws[k] * (1.0f / (float)N_TOK);     // avg_probs[k]
    red[k] = a * __logf(a + 1e-10f);
    __syncthreads();
    for (int s = K_CODE / 2; s > 0; s >>= 1) {
        if (k < s) red[k] += red[k + s];
        __syncthreads();
    }
    if (k == 0) {
        out[P_OFF] = __expf(-red[0]);                      // perplexity
        out[0]     = ws[K_CODE] * (1.0f / (float)K_CODE);  // KL = sum/K
    }
}

extern "C" void kernel_launch(void* const* d_in, const int* in_sizes, int n_in,
                              void* d_out, int out_size, void* d_ws, size_t ws_size,
                              hipStream_t stream)
{
    const float* x   = (const float*)d_in[0]; // inputs  [64,2048,128]
    const float* u   = (const float*)d_in[1]; // uniform [131072,512]
    const float* emb = (const float*)d_in[2]; // codebook [512,128]
    float* out = (float*)d_out;
    float* ws  = (float*)d_ws;

    // workspace: ws[0..511] = sum_enc per code, ws[512] = raw KL sum
    (void)hipMemsetAsync(d_ws, 0, (K_CODE + 1) * sizeof(float), stream);

    vq_fused_kernel<<<GRID_BLKS, BLK, 0, stream>>>(x, u, emb, out, ws);
    vq_finalize_kernel<<<1, K_CODE, 0, stream>>>(ws, out);
}
